// RelGraphConv_67302137528493
// MI455X (gfx1250) — matrix operand order repaired and checked by reading, not compiled, not run on hardware
//
#include <hip/hip_runtime.h>
#include <hip/hip_bf16.h>

#define N_NODES 40000
#define N_EDGES 400000
#define FDIM 64
#define RREL 20
#define HEADS 3

typedef __bf16 bf16_t;
typedef __attribute__((ext_vector_type(16))) __bf16 v16bf;
typedef __attribute__((ext_vector_type(8)))  __bf16 v8bf;
typedef __attribute__((ext_vector_type(2)))  __bf16 v2bf;
typedef __attribute__((ext_vector_type(8)))  float  v8f;
typedef int v4i_gcc __attribute__((vector_size(4 * sizeof(int))));

#if defined(__has_builtin)
#if __has_builtin(__builtin_amdgcn_global_load_async_to_lds_b128)
#define HAVE_ASYNC_LDS 1
#endif
#if __has_builtin(__builtin_amdgcn_s_wait_asynccnt)
#define HAVE_WAIT_ASYNC 1
#endif
#endif

// ---------------- helpers ----------------

__device__ inline unsigned fenc(float x) {
  unsigned u = __float_as_uint(x);
  return (u & 0x80000000u) ? ~u : (u | 0x80000000u);
}
__device__ inline float fdec(unsigned e) {
  unsigned u = (e & 0x80000000u) ? (e & 0x7FFFFFFFu) : ~e;
  return __uint_as_float(u);
}
#define ENC_NEG_INF 0x007FFFFFu

__device__ inline void atomAddF(float* p, float v) {
#if defined(__gfx1250__)
  unsafeAtomicAdd(p, v);     // native global_atomic_add_f32
#else
  atomicAdd(p, v);
#endif
}

__device__ inline float leaky(float x) { return x > 0.f ? x : 0.2f * x; }
__device__ inline float sigmoidf(float x) { return 1.f / (1.f + expf(-x)); }

__device__ inline v8f wmma_bf16(v16bf a, v16bf b, v8f c) {
  return __builtin_amdgcn_wmma_f32_16x16x32_bf16(false, a, false, b, (short)0, c,
                                                 false, false);
}

// A fragment from row-major LDS tile A[m][k] (stride mult of 8, 16B-aligned base).
// Lane layout (ISA 7.12.2): m = lane&15, kb = (lane>=16)?8:0;
// halves 0..7  = K=kb+0..7   (contiguous 16B)
// halves 8..15 = K=kb+16..23 (contiguous 16B)
__device__ inline v16bf load_a_tile(const bf16_t* A, int stride, int k0, int lane) {
  int m  = lane & 15;
  int kb = (lane >> 4) * 8;
  const bf16_t* p = A + m * stride + k0 + kb;
  v16bf a;
  *(v8bf*)&a       = *(const v8bf*)p;
  *((v8bf*)&a + 1) = *(const v8bf*)(p + 16);
  return a;
}
// B fragment from TRANSPOSED LDS tile Bt[n][k] (k contiguous, strideK mult of 8).
// Lane layout: n = lane&15, halves 0..7 = K=kb+0..7, halves 8..15 = K=kb+16..23 with kb = (lane>=16)?8:0.
__device__ inline v16bf load_b_tile_t(const bf16_t* Bt, int strideK, int k0, int lane) {
  int n  = lane & 15;
  int kb = (lane >> 4) * 8;
  const bf16_t* p = Bt + n * strideK + k0 + kb;
  v16bf b;
  *(v8bf*)&b       = *(const v8bf*)p;
  *((v8bf*)&b + 1) = *(const v8bf*)(p + 16);
  return b;
}

#ifdef HAVE_ASYNC_LDS
// Per diagnostic: param0 is addrspace(1) pointer to int __attribute__((vector_size(16))).
__device__ inline void async_cp16(const void* g, void* l) {
  __builtin_amdgcn_global_load_async_to_lds_b128(
      (__attribute__((address_space(1))) v4i_gcc*)g,
      (__attribute__((address_space(3))) v4i_gcc*)l, 0, 0);
}
#endif
__device__ inline void wait_async0() {
#ifdef HAVE_WAIT_ASYNC
  __builtin_amdgcn_s_wait_asynccnt(0);
#else
  asm volatile("s_wait_asynccnt 0" ::: "memory");
#endif
}

// ---------------- kernels ----------------

__global__ void k_f2bf(const float* __restrict__ x, bf16_t* __restrict__ y, int n) {
  int i = blockIdx.x * blockDim.x + threadIdx.x;
  if (i < n) y[i] = (bf16_t)x[i];
}

// Convert + transpose: x is [K][N] row-major f32, y is [N][K] bf16.
__global__ void k_f2bf_t(const float* __restrict__ x, bf16_t* __restrict__ y,
                         int K, int N) {
  int i = blockIdx.x * blockDim.x + threadIdx.x;
  if (i >= K * N) return;
  int k = i / N, n = i % N;
  y[n * K + k] = (bf16_t)x[i];
}

// Per-head tiny precompute: u_l/u_r[R*F], w_l/w_r[F], c (single block).
__global__ void k_head_consts(const float* __restrict__ Watt, const float* __restrict__ avec,
                              const float* __restrict__ loop_atten,
                              const float* __restrict__ Wlin, const float* __restrict__ blin,
                              int kh, float* __restrict__ consts) {
  __shared__ float vl[FDIM], vr[FDIM];
  const float* al = avec + kh * 2 * FDIM;
  const float* ar = al + FDIM;
  int t = threadIdx.x;
  if (t < FDIM) {
    float s = 0.f;
    for (int g = 0; g < FDIM; ++g) s += loop_atten[t * FDIM + g] * al[g];
    vl[t] = s;
  } else if (t < 2 * FDIM) {
    int f = t - FDIM; float s = 0.f;
    for (int g = 0; g < FDIM; ++g) s += loop_atten[f * FDIM + g] * ar[g];
    vr[f] = s;
  }
  __syncthreads();
  float* ul = consts;
  float* ur = consts + RREL * FDIM;
  float* wl = ur + RREL * FDIM;
  float* wr = wl + FDIM;
  float* cc = wr + FDIM;
  const float* WA = Watt + (size_t)kh * RREL * FDIM * FDIM;
  for (int idx = t; idx < 2 * RREL * FDIM; idx += blockDim.x) {
    int half = idx / (RREL * FDIM);
    int rf   = idx % (RREL * FDIM);
    const float* row = WA + (size_t)rf * FDIM;
    const float* v   = half ? vr : vl;
    float s = 0.f;
    for (int g = 0; g < FDIM; ++g) s += row[g] * v[g];
    (half ? ur : ul)[rf] = s;
  }
  if (t < FDIM) {
    float s = 0.f;
    for (int g = 0; g < FDIM; ++g) s += Wlin[t * FDIM + g] * al[g];
    wl[t] = s;
  } else if (t < 2 * FDIM) {
    int f = t - FDIM; float s = 0.f;
    for (int g = 0; g < FDIM; ++g) s += Wlin[f * FDIM + g] * ar[g];
    wr[f] = s;
  }
  if (t == 0) {
    float s = 0.f;
    for (int g = 0; g < FDIM; ++g) s += blin[g] * (al[g] + ar[g]);
    *cc = s;
  }
}

// Per-node dot products: s_l[r][n], s_r[r][n], t_l[n], t_r[n].
__global__ void k_node_dots(const float* __restrict__ feat, const float* __restrict__ consts,
                            float* __restrict__ sl, float* __restrict__ sr,
                            float* __restrict__ tl, float* __restrict__ tr) {
  __shared__ float ul[RREL * FDIM], ur[RREL * FDIM], wl[FDIM], wr[FDIM];
  for (int i = threadIdx.x; i < RREL * FDIM; i += blockDim.x) {
    ul[i] = consts[i];
    ur[i] = consts[RREL * FDIM + i];
  }
  for (int i = threadIdx.x; i < FDIM; i += blockDim.x) {
    wl[i] = consts[2 * RREL * FDIM + i];
    wr[i] = consts[2 * RREL * FDIM + FDIM + i];
  }
  __syncthreads();
  int n = blockIdx.x * blockDim.x + threadIdx.x;
  if (n >= N_NODES) return;
  float f[FDIM];
#pragma unroll
  for (int g = 0; g < FDIM; ++g) f[g] = feat[(size_t)n * FDIM + g];
  float a = 0.f, b = 0.f;
#pragma unroll
  for (int g = 0; g < FDIM; ++g) { a += f[g] * wl[g]; b += f[g] * wr[g]; }
  tl[n] = a; tr[n] = b;
  for (int r = 0; r < RREL; ++r) {
    float s1 = 0.f, s2 = 0.f;
#pragma unroll
    for (int g = 0; g < FDIM; ++g) {
      s1 += f[g] * ul[r * FDIM + g];
      s2 += f[g] * ur[r * FDIM + g];
    }
    sl[(size_t)r * N_NODES + n] = s1;
    sr[(size_t)r * N_NODES + n] = s2;
  }
}

__global__ void k_fill_head(unsigned* __restrict__ dmax, float* __restrict__ dsum,
                            unsigned* __restrict__ gmax, float* __restrict__ gsum) {
  int i = blockIdx.x * blockDim.x + threadIdx.x;
  if (i < N_NODES) { dmax[i] = ENC_NEG_INF; dsum[i] = 0.f; }
  if (i == 0) { *gmax = ENC_NEG_INF; *gsum = 0.f; }
}

// e1 masked + global max
__global__ void k_e1(const float* __restrict__ sl, const float* __restrict__ sr,
                     const int* __restrict__ src, const int* __restrict__ dst,
                     const int* __restrict__ ety,
                     float* __restrict__ ebuf, unsigned* __restrict__ gmax) {
  int e = blockIdx.x * blockDim.x + threadIdx.x;
  float v = -3.402823e38f;
  if (e < N_EDGES) {
    int et = ety[e];
    float m = 0.f;
    if (et == 19) {
      float x = sl[(size_t)et * N_NODES + src[e]] + sr[(size_t)et * N_NODES + dst[e]];
      m = leaky(x);
    }
    ebuf[e] = m;
    v = m;
  }
  for (int o = 16; o > 0; o >>= 1) v = fmaxf(v, __shfl_xor(v, o, 32));
  __shared__ float wred[8];
  int w = threadIdx.x >> 5;
  if ((threadIdx.x & 31) == 0) wred[w] = v;
  __syncthreads();
  if (threadIdx.x == 0) {
    float m = wred[0];
    for (int i = 1; i < (int)(blockDim.x >> 5); ++i) m = fmaxf(m, wred[i]);
    atomicMax(gmax, fenc(m));
  }
}

// exp(e1 - gmax), global sum
__global__ void k_e2exp(const float* __restrict__ ein, const unsigned* __restrict__ gmax,
                        float* __restrict__ eout, float* __restrict__ gsum) {
  int e = blockIdx.x * blockDim.x + threadIdx.x;
  float gm = fdec(*gmax);
  float ex = 0.f;
  if (e < N_EDGES) { ex = expf(ein[e] - gm); eout[e] = ex; }
  for (int o = 16; o > 0; o >>= 1) ex += __shfl_xor(ex, o, 32);
  __shared__ float wred[8];
  int w = threadIdx.x >> 5;
  if ((threadIdx.x & 31) == 0) wred[w] = ex;
  __syncthreads();
  if (threadIdx.x == 0) {
    float s = 0.f;
    for (int i = 0; i < (int)(blockDim.x >> 5); ++i) s += wred[i];
    atomAddF(gsum, s);
  }
}

// e3 = leaky(e2*(t_l[src]+t_r[dst]) + c); segment max over dst
__global__ void k_e3(const float* __restrict__ exbuf, const float* __restrict__ gsum,
                     const float* __restrict__ tl, const float* __restrict__ tr,
                     const float* __restrict__ consts,
                     const int* __restrict__ src, const int* __restrict__ dst,
                     float* __restrict__ e3buf, unsigned* __restrict__ dmax) {
  int e = blockIdx.x * blockDim.x + threadIdx.x;
  if (e >= N_EDGES) return;
  float c  = consts[2 * RREL * FDIM + 2 * FDIM];
  float e2 = exbuf[e] / (*gsum);
  int s = src[e], d = dst[e];
  float x = e2 * (tl[s] + tr[d]) + c;
  float e3 = leaky(x);
  e3buf[e] = e3;
  atomicMax(&dmax[d], fenc(e3));
}

// ex3 = exp(e3 - dmax[dst]); segment sum over dst
__global__ void k_exp3(const float* __restrict__ e3buf, const unsigned* __restrict__ dmax,
                       const int* __restrict__ dst,
                       float* __restrict__ exbuf, float* __restrict__ dsum) {
  int e = blockIdx.x * blockDim.x + threadIdx.x;
  if (e >= N_EDGES) return;
  int d = dst[e];
  float ex = expf(e3buf[e] - fdec(dmax[d]));
  exbuf[e] = ex;
  atomAddF(&dsum[d], ex);
}

// Edge aggregation: one wave per edge; m_src = feat[src]@Wr[et] (Wr bf16 in LDS),
// lb_edge = sigmoid(label[src]@W_lb[et]); scatter-add bs/ws with float atomics.
__global__ void k_aggregate(const float* __restrict__ feat, const float* __restrict__ label,
                            const bf16_t* __restrict__ WrBf, const float* __restrict__ Wlb,
                            const int* __restrict__ src, const int* __restrict__ dst,
                            const int* __restrict__ ety,
                            const float* __restrict__ exbuf, const float* __restrict__ dsum,
                            float* __restrict__ bs, float* __restrict__ wsv) {
  extern __shared__ char smem[];
  float*  lWlb = (float*)smem;                               // R*2*F floats (10KB)
  bf16_t* lWr  = (bf16_t*)(smem + RREL * 2 * FDIM * 4);      // R*F*F bf16 (160KB)
  {  // vectorized LDS fills (16B chunks)
    const uint4* gw = (const uint4*)Wlb;
    uint4* sw = (uint4*)lWlb;
    for (int i = threadIdx.x; i < RREL * 2 * FDIM / 4; i += blockDim.x) sw[i] = gw[i];
    const uint4* gr = (const uint4*)WrBf;
    uint4* sr = (uint4*)lWr;
    for (int i = threadIdx.x; i < RREL * FDIM * FDIM / 8; i += blockDim.x) sr[i] = gr[i];
  }
  __syncthreads();
  int lane  = threadIdx.x & 31;
  int warp  = blockIdx.x * (blockDim.x >> 5) + (threadIdx.x >> 5);
  int nwarp = gridDim.x * (blockDim.x >> 5);
  int f0 = lane * 2;
  for (int e = warp; e < N_EDGES; e += nwarp) {
    if (e + nwarp < N_EDGES) {
      __builtin_prefetch(&src[e + nwarp], 0, 0);   // global_prefetch_b8
      __builtin_prefetch(&dst[e + nwarp], 0, 0);
    }
    int s = src[e], d = dst[e], et = ety[e];
    float l0 = label[2 * s], l1 = label[2 * s + 1];
    float blk = (l1 >= 0.5f) ? 1.f : 0.f;
    float wht = (l0 >= 0.5f) ? 1.f : 0.f;
    if (blk == 0.f && wht == 0.f) continue;
    float att = exbuf[e] / (dsum[d] + 1e-16f);
    float2 fr = *(const float2*)&feat[(size_t)s * FDIM + f0];
    const bf16_t* wr = lWr + et * FDIM * FDIM;
    float m0 = 0.f, m1 = 0.f;
#pragma unroll
    for (int g = 0; g < FDIM; ++g) {
      float fg = __shfl((g & 1) ? fr.y : fr.x, g >> 1, 32);
      v2bf w = *(const v2bf*)(wr + g * FDIM + f0);   // one ds_load_b32
      m0 += fg * (float)w[0];
      m1 += fg * (float)w[1];
    }
    const float* wlb = lWlb + et * 2 * FDIM;
    float lb0 = sigmoidf(l0 * wlb[f0]     + l1 * wlb[FDIM + f0]);
    float lb1 = sigmoidf(l0 * wlb[f0 + 1] + l1 * wlb[FDIM + f0 + 1]);
    float md0 = lb0 * m0 * att;
    float md1 = lb1 * m1 * att;
    if (blk != 0.f) {
      atomAddF(&bs[(size_t)d * FDIM + f0],     md0);
      atomAddF(&bs[(size_t)d * FDIM + f0 + 1], md1);
    }
    if (wht != 0.f) {
      atomAddF(&wsv[(size_t)d * FDIM + f0],     md0);
      atomAddF(&wsv[(size_t)d * FDIM + f0 + 1], md1);
    }
  }
}

// Gate MLP via WMMA: x=[bs|ws] (16x128) -> relu(x@Wg1+bg1) (16x256) -> sigmoid(@Wg2+bg2);
// h_acc += gate*bs + (1-gate)*ws. 128 threads = 4 waves per 16-row block.
// Weights are pre-transposed ([N][K], k-contiguous) for direct B-fragment loads.
__global__ void k_gate(const float* __restrict__ bs, const float* __restrict__ wsv,
                       const bf16_t* __restrict__ Wg1t, const bf16_t* __restrict__ Wg2t,
                       const float* __restrict__ bg1, const float* __restrict__ bg2,
                       float* __restrict__ hacc) {
  extern __shared__ char smem[];
  bf16_t* lW1 = (bf16_t*)smem;            // [256][128] transposed Wg1
  bf16_t* lW2 = lW1 + 256 * 128;          // [64][256]  transposed Wg2
  bf16_t* lA  = lW2 + 64 * 256;           // [16][128]
  bf16_t* lH  = lA + 16 * 128;            // [16][256]
  int tid = threadIdx.x;
#ifdef HAVE_ASYNC_LDS
  {
    const char* g1 = (const char*)Wg1t;
    char* l1 = (char*)lW1;
    for (int i = tid * 16; i < 256 * 128 * 2; i += (int)blockDim.x * 16)
      async_cp16(g1 + i, l1 + i);
    const char* g2 = (const char*)Wg2t;
    char* l2 = (char*)lW2;
    for (int i = tid * 16; i < 64 * 256 * 2; i += (int)blockDim.x * 16)
      async_cp16(g2 + i, l2 + i);
  }
#else
  {
    const uint4* g1 = (const uint4*)Wg1t;
    uint4* s1 = (uint4*)lW1;
    for (int i = tid; i < 256 * 128 / 8; i += blockDim.x) s1[i] = g1[i];
    const uint4* g2 = (const uint4*)Wg2t;
    uint4* s2 = (uint4*)lW2;
    for (int i = tid; i < 64 * 256 / 8; i += blockDim.x) s2[i] = g2[i];
  }
#endif
  int m0 = blockIdx.x * 16;
  for (int i = tid * 2; i < 16 * 128; i += (int)blockDim.x * 2) {
    int m = i >> 7, k = i & 127;
    const float* row = (k < 64) ? &bs[(size_t)(m0 + m) * FDIM + k]
                                : &wsv[(size_t)(m0 + m) * FDIM + (k - 64)];
    v2bf p;
    p[0] = (bf16_t)row[0];
    p[1] = (bf16_t)row[1];
    *(v2bf*)&lA[i] = p;
  }
#ifdef HAVE_ASYNC_LDS
  wait_async0();
#endif
  __syncthreads();
  int wv = tid >> 5, lane = tid & 31;
  // GEMM1: 16x128 @ 128x256, 4 col-tiles per wave
  for (int ti = 0; ti < 4; ++ti) {
    int ct = wv * 4 + ti;
    v8f acc = {};
#pragma unroll
    for (int k0 = 0; k0 < 128; k0 += 32)
      acc = wmma_bf16(load_a_tile(lA, 128, k0, lane),
                      load_b_tile_t(lW1 + ct * 16 * 128, 128, k0, lane), acc);
    int nn = ct * 16 + (lane & 15);
    int mb = (lane >> 4) * 8;
    float bg = bg1[nn];
#pragma unroll
    for (int g = 0; g < 8; ++g) {
      float h = acc[g] + bg;
      h = (h > 0.f) ? h : 0.f;
      lH[(mb + g) * 256 + nn] = (bf16_t)h;
    }
  }
  __syncthreads();
  // GEMM2: 16x256 @ 256x64, 1 col-tile per wave
  {
    int ct = wv;
    v8f acc = {};
#pragma unroll
    for (int k0 = 0; k0 < 256; k0 += 32)
      acc = wmma_bf16(load_a_tile(lH, 256, k0, lane),
                      load_b_tile_t(lW2 + ct * 16 * 256, 256, k0, lane), acc);
    int nn = ct * 16 + (lane & 15);
    int mb = (lane >> 4) * 8;
    float bg = bg2[nn];
#pragma unroll
    for (int g = 0; g < 8; ++g) {
      int node = m0 + mb + g;
      float gate = sigmoidf(acc[g] + bg);
      float b_ = bs[(size_t)node * FDIM + nn];
      float w_ = wsv[(size_t)node * FDIM + nn];
      hacc[(size_t)node * FDIM + nn] += gate * b_ + (1.f - gate) * w_;
    }
  }
}

// out_pre = (h_acc/3)@W_lin + feat@loop_weight + (b_lin + h_bias), WMMA.
__global__ void k_out(const float* __restrict__ hacc, const bf16_t* __restrict__ featbf,
                      const bf16_t* __restrict__ WlinT, const bf16_t* __restrict__ LwT,
                      const float* __restrict__ blin, const float* __restrict__ hbias,
                      float* __restrict__ out) {
  __shared__ __align__(16) bf16_t lA1[16 * 64];
  __shared__ __align__(16) bf16_t lA2[16 * 64];
  __shared__ __align__(16) bf16_t lW[64 * 64];   // transposed W_lin [n][k]
  __shared__ __align__(16) bf16_t lL[64 * 64];   // transposed loop_weight [n][k]
  int tid = threadIdx.x;
  int m0 = blockIdx.x * 16;
  for (int i = tid * 2; i < 16 * 64; i += (int)blockDim.x * 2) {
    int m = i >> 6, k = i & 63;
    v2bf p;
    p[0] = (bf16_t)(hacc[(size_t)(m0 + m) * FDIM + k]     * (1.f / 3.f));
    p[1] = (bf16_t)(hacc[(size_t)(m0 + m) * FDIM + k + 1] * (1.f / 3.f));
    *(v2bf*)&lA1[i] = p;
    *(v2bf*)&lA2[i] = *(const v2bf*)&featbf[(size_t)(m0 + m) * FDIM + k];
  }
  {
    const uint4* gw = (const uint4*)WlinT;
    const uint4* gl = (const uint4*)LwT;
    uint4* sw = (uint4*)lW;
    uint4* sl2 = (uint4*)lL;
    for (int i = tid; i < 64 * 64 / 8; i += blockDim.x) { sw[i] = gw[i]; sl2[i] = gl[i]; }
  }
  __syncthreads();
  int wv = tid >> 5, lane = tid & 31;
  int ct = wv;
  v8f acc = {};
  acc = wmma_bf16(load_a_tile(lA1, 64, 0,  lane), load_b_tile_t(lW + ct * 16 * 64, 64, 0,  lane), acc);
  acc = wmma_bf16(load_a_tile(lA1, 64, 32, lane), load_b_tile_t(lW + ct * 16 * 64, 64, 32, lane), acc);
  acc = wmma_bf16(load_a_tile(lA2, 64, 0,  lane), load_b_tile_t(lL + ct * 16 * 64, 64, 0,  lane), acc);
  acc = wmma_bf16(load_a_tile(lA2, 64, 32, lane), load_b_tile_t(lL + ct * 16 * 64, 64, 32, lane), acc);
  int nn = ct * 16 + (lane & 15);
  int mb = (lane >> 4) * 8;
  float bias = blin[nn] + hbias[nn];
#pragma unroll
  for (int g = 0; g < 8; ++g)
    out[(size_t)(m0 + mb + g) * FDIM + nn] = acc[g] + bias;
}

// In-place LayerNorm over F=64; one wave per row, 2 cols/lane.
__global__ void k_ln(float* __restrict__ out, const float* __restrict__ gamma,
                     const float* __restrict__ beta) {
  int lane  = threadIdx.x & 31;
  int warp  = blockIdx.x * (blockDim.x >> 5) + (threadIdx.x >> 5);
  int nwarp = gridDim.x * (blockDim.x >> 5);
  for (int r = warp; r < N_NODES; r += nwarp) {
    float x0 = out[(size_t)r * FDIM + lane * 2];
    float x1 = out[(size_t)r * FDIM + lane * 2 + 1];
    float s = x0 + x1;
    for (int o = 16; o > 0; o >>= 1) s += __shfl_xor(s, o, 32);
    float mu = s * (1.f / 64.f);
    float d0 = x0 - mu, d1 = x1 - mu;
    float v = d0 * d0 + d1 * d1;
    for (int o = 16; o > 0; o >>= 1) v += __shfl_xor(v, o, 32);
    float inv = rsqrtf(v * (1.f / 64.f) + 1e-5f);
    out[(size_t)r * FDIM + lane * 2]     = d0 * inv * gamma[lane * 2]     + beta[lane * 2];
    out[(size_t)r * FDIM + lane * 2 + 1] = d1 * inv * gamma[lane * 2 + 1] + beta[lane * 2 + 1];
  }
}

// ---------------- host ----------------

extern "C" void kernel_launch(void* const* d_in, const int* in_sizes, int n_in,
                              void* d_out, int out_size, void* d_ws, size_t ws_size,
                              hipStream_t stream) {
  const float* feat  = (const float*)d_in[0];
  const float* label = (const float*)d_in[1];
  const float* Watt  = (const float*)d_in[2];
  const float* avec  = (const float*)d_in[3];
  const float* loopa = (const float*)d_in[4];
  const float* Wlin  = (const float*)d_in[5];
  const float* blin  = (const float*)d_in[6];
  const float* Wr    = (const float*)d_in[7];
  const float* Wlb   = (const float*)d_in[8];
  const float* Wg1   = (const float*)d_in[9];
  const float* bg1   = (const float*)d_in[10];
  const float* Wg2   = (const float*)d_in[11];
  const float* bg2   = (const float*)d_in[12];
  const float* loopw = (const float*)d_in[13];
  const float* hbias = (const float*)d_in[14];
  const float* lng   = (const float*)d_in[15];
  const float* lnb   = (const float*)d_in[16];
  const int*   src   = (const int*)d_in[17];
  const int*   dst   = (const int*)d_in[18];
  const int*   ety   = (const int*)d_in[19];

  char* ws = (char*)d_ws;
  size_t off = 0;
  auto carve = [&](size_t bytes) -> char* {
    char* p = ws + off;
    off = (off + bytes + 255) & ~(size_t)255;
    return p;
  };
  bf16_t* featbf = (bf16_t*)carve((size_t)N_NODES * FDIM * 2);
  bf16_t* wrbf   = (bf16_t*)carve((size_t)RREL * FDIM * FDIM * 2);
  bf16_t* wg1t   = (bf16_t*)carve((size_t)2 * FDIM * 4 * FDIM * 2);
  bf16_t* wg2t   = (bf16_t*)carve((size_t)4 * FDIM * FDIM * 2);
  bf16_t* wlinT  = (bf16_t*)carve((size_t)FDIM * FDIM * 2);
  bf16_t* lwT    = (bf16_t*)carve((size_t)FDIM * FDIM * 2);
  float*  consts = (float*)carve((size_t)(2 * RREL * FDIM + 2 * FDIM + 4) * 4);
  float*  sl     = (float*)carve((size_t)RREL * N_NODES * 4);
  float*  sr     = (float*)carve((size_t)RREL * N_NODES * 4);
  float*  tl     = (float*)carve((size_t)N_NODES * 4);
  float*  tr     = (float*)carve((size_t)N_NODES * 4);
  float*  ebufA  = (float*)carve((size_t)N_EDGES * 4);
  float*  ebufB  = (float*)carve((size_t)N_EDGES * 4);
  unsigned* gmax = (unsigned*)carve(16);
  float*  gsum   = (float*)((char*)gmax + 8);
  unsigned* dmax = (unsigned*)carve((size_t)N_NODES * 4);
  float*  dsum   = (float*)carve((size_t)N_NODES * 4);
  float*  bsb    = (float*)carve((size_t)N_NODES * FDIM * 4);
  float*  wsb    = (float*)carve((size_t)N_NODES * FDIM * 4);
  float*  hacc   = (float*)carve((size_t)N_NODES * FDIM * 4);

  const int NF = N_NODES * FDIM;
  (void)hipMemsetAsync(hacc, 0, (size_t)NF * 4, stream);

  k_f2bf<<<(NF + 255) / 256, 256, 0, stream>>>(feat, featbf, NF);
  k_f2bf<<<(RREL * FDIM * FDIM + 255) / 256, 256, 0, stream>>>(Wr, wrbf, RREL * FDIM * FDIM);
  k_f2bf_t<<<(2 * FDIM * 4 * FDIM + 255) / 256, 256, 0, stream>>>(Wg1, wg1t, 2 * FDIM, 4 * FDIM);
  k_f2bf_t<<<(4 * FDIM * FDIM + 255) / 256, 256, 0, stream>>>(Wg2, wg2t, 4 * FDIM, FDIM);
  k_f2bf_t<<<(FDIM * FDIM + 255) / 256, 256, 0, stream>>>(Wlin, wlinT, FDIM, FDIM);
  k_f2bf_t<<<(FDIM * FDIM + 255) / 256, 256, 0, stream>>>(loopw, lwT, FDIM, FDIM);

  const int EB = (N_EDGES + 255) / 256;
  for (int kh = 0; kh < HEADS; ++kh) {
    k_head_consts<<<1, 256, 0, stream>>>(Watt, avec, loopa, Wlin, blin, kh, consts);
    k_node_dots<<<(N_NODES + 127) / 128, 128, 0, stream>>>(feat, consts, sl, sr, tl, tr);
    k_fill_head<<<(N_NODES + 255) / 256, 256, 0, stream>>>(dmax, dsum, gmax, gsum);
    k_e1<<<EB, 256, 0, stream>>>(sl, sr, src, dst, ety, ebufA, gmax);
    k_e2exp<<<EB, 256, 0, stream>>>(ebufA, gmax, ebufB, gsum);
    k_e3<<<EB, 256, 0, stream>>>(ebufB, gsum, tl, tr, consts, src, dst, ebufA, dmax);
    k_exp3<<<EB, 256, 0, stream>>>(ebufA, dmax, dst, ebufB, dsum);
    (void)hipMemsetAsync(bsb, 0, (size_t)NF * 4, stream);
    (void)hipMemsetAsync(wsb, 0, (size_t)NF * 4, stream);
    size_t aggLds = (size_t)RREL * 2 * FDIM * 4 + (size_t)RREL * FDIM * FDIM * 2; // 174080
    k_aggregate<<<512, 256, aggLds, stream>>>(feat, label, wrbf, Wlb, src, dst, ety,
                                              ebufB, dsum, bsb, wsb);
    size_t gateLds = (size_t)(256 * 128 + 64 * 256 + 16 * 128 + 16 * 256) * 2; // 110592
    k_gate<<<N_NODES / 16, 128, gateLds, stream>>>(bsb, wsb, wg1t, wg2t, bg1, bg2, hacc);
  }
  k_out<<<N_NODES / 16, 128, 0, stream>>>(hacc, featbf, wlinT, lwT, blin, hbias,
                                          (float*)d_out);
  k_ln<<<256, 256, 0, stream>>>((float*)d_out, lng, lnb);
}
